// LocalSemanticField_6682969112753
// MI455X (gfx1250) — matrix operand order repaired and checked by reading, not compiled, not run on hardware
//
#include <hip/hip_runtime.h>
#include <hip/hip_fp16.h>
#include <math.h>

typedef __attribute__((ext_vector_type(16))) _Float16 v16h;
typedef __attribute__((ext_vector_type(8)))  _Float16 v8h;
typedef __attribute__((ext_vector_type(8)))  float    v8f;

#define BB 8
#define SS 2048
#define PP 2048
#define DD 256
#define PSTRIDE 16    // center, phase, q0..3, w0..3, cf0..3, pad x2  (64B records)
#define KSTEP 32
#define BLK_P 64      // block covers 64 P x full 256 D -> basis computed exactly once
#define A_STRIDE 40   // halfs per p-row in LDS: 32 + 8 pad (80B, keeps 16B alignment)

// ---------------- conv stack (one-hot conv1 is a gather) ----------------

__global__ void k_conv1(const int* __restrict__ bytes, const float* __restrict__ w1,
                        const float* __restrict__ b1, float* __restrict__ h1) {
    int t = blockIdx.x * blockDim.x + threadIdx.x;      // B*128*S threads
    int s = t % SS; int c = (t / SS) & 127; int b = t / (SS * 128);
    float acc = b1[c];
#pragma unroll
    for (int k = 0; k < 7; ++k) {
        int sp = s + k - 3;
        if (sp >= 0 && sp < SS) {
            int idx = bytes[b * SS + sp];
            acc += w1[(c * 256 + idx) * 7 + k];
        }
    }
    float g = 0.5f * acc * (1.0f + erff(acc * 0.70710678118f));   // exact GELU
    h1[((size_t)(b * 128 + c)) * SS + s] = g;
}

__global__ void k_conv2(const float* __restrict__ h1, const float* __restrict__ w2,
                        const float* __restrict__ b2, float* __restrict__ h2) {
    int t = blockIdx.x * blockDim.x + threadIdx.x;      // B*64*S threads
    int s = t % SS; int c2 = (t / SS) & 63; int b = t / (SS * 64);
    float acc = b2[c2];
    for (int c1 = 0; c1 < 128; ++c1) {
        const float* hrow = h1 + ((size_t)(b * 128 + c1)) * SS;
        const float* wrow = w2 + (c2 * 128 + c1) * 5;
#pragma unroll
        for (int k = 0; k < 5; ++k) {
            int sp = s + k - 2;
            float hv = (sp >= 0 && sp < SS) ? hrow[sp] : 0.0f;
            acc = fmaf(wrow[k], hv, acc);
        }
    }
    float g = 0.5f * acc * (1.0f + erff(acc * 0.70710678118f));
    h2[((size_t)(b * 64 + c2)) * SS + s] = g;
}

__global__ void k_conv3(const float* __restrict__ h2, const float* __restrict__ w3,
                        const float* __restrict__ b3, float* __restrict__ ctx) {
    int t = blockIdx.x * blockDim.x + threadIdx.x;      // B*S threads
    int s = t % SS; int b = t / SS;
    float acc = b3[0];
    for (int c = 0; c < 64; ++c) {
        const float* hrow = h2 + ((size_t)(b * 64 + c)) * SS;
#pragma unroll
        for (int k = 0; k < 3; ++k) {
            int sp = s + k - 1;
            float hv = (sp >= 0 && sp < SS) ? hrow[sp] : 0.0f;
            acc = fmaf(w3[c * 3 + k], hv, acc);
        }
    }
    ctx[t] = 1.0f / (1.0f + __expf(-acc));              // sigmoid
}

// --- per-(b,s) params: center, phase, q=-0.5/width^2, w=2pi/width, cf=amp_scale*sw^2 ---

__global__ void k_params(const int* __restrict__ bytes, const float* __restrict__ lsh,
                         const float* __restrict__ msw, const float* __restrict__ sw,
                         float* __restrict__ prm) {
    int t = blockIdx.x * blockDim.x + threadIdx.x;      // B*S threads
    int s = t % SS;
    int idx = bytes[t];
    const float* l = lsh + idx * 4;
    float center = (float)s / (float)(SS - 1) + l[0] * 0.05f;
    float absw   = fabsf(l[1]);
    float ampsc  = 1.0f / (1.0f + __expf(-l[2]));
    float phase  = l[3] * 3.14159265358979f;
    float* o = prm + (size_t)t * PSTRIDE;
    o[0] = center; o[1] = phase;
#pragma unroll
    for (int sc = 0; sc < 4; ++sc) {
        float iw = 1.0f / (absw * msw[sc] + 1e-5f);
        o[2 + sc]  = -0.5f * iw * iw;                   // q
        o[6 + sc]  = 6.28318530718f * iw;               // w
        o[10 + sc] = ampsc * sw[sc] * sw[sc];           // cf
    }
    o[14] = 0.0f; o[15] = 0.0f;
}

// ------------- amps = emb[byte]*ctx, transposed to [B][D][S] fp16 -------------

__global__ void k_amps(const int* __restrict__ bytes, const float* __restrict__ emb,
                       const float* __restrict__ ctx, _Float16* __restrict__ ampsT) {
    int t = blockIdx.x * blockDim.x + threadIdx.x;      // B*D*S threads, s fastest
    int s = t % SS; int d = (t / SS) & (DD - 1); int b = t / (SS * DD);
    int idx = bytes[b * SS + s];
    float v = emb[idx * DD + d] * ctx[b * SS + s];
    ampsT[((size_t)(b * DD + d)) * SS + s] = (_Float16)v;
}

// ------------- fused basis-generation + WMMA GEMM: out[b][p][d] -------------

__device__ __forceinline__ void gen_A(_Float16* buf, const float* __restrict__ prm_base,
                                      const float* pos_sh, int k0, int kk, int pr) {
    const float* prm = prm_base + (size_t)(k0 + kk) * PSTRIDE;
    const float center = prm[0], phase = prm[1];
    const float q0 = prm[2], q1 = prm[3], q2 = prm[4], q3 = prm[5];
    const float w0 = prm[6], w1 = prm[7], w2 = prm[8], w3 = prm[9];
    const float c0 = prm[10], c1 = prm[11], c2 = prm[12], c3 = prm[13];
#pragma unroll
    for (int i = 0; i < 8; ++i) {
        int p = pr * 8 + i;
        float xd = pos_sh[p] - center;
        float x2 = xd * xd;
        float v;
        v = c0 * __expf(q0 * x2) * __cosf(fmaf(w0, xd, phase));
        v = fmaf(c1 * __expf(q1 * x2), __cosf(fmaf(w1, xd, phase)), v);
        v = fmaf(c2 * __expf(q2 * x2), __cosf(fmaf(w2, xd, phase)), v);
        v = fmaf(c3 * __expf(q3 * x2), __cosf(fmaf(w3, xd, phase)), v);
        buf[p * A_STRIDE + kk] = (_Float16)v;
    }
}

__global__ __launch_bounds__(256) void k_field(const float* __restrict__ pos,
                                               const float* __restrict__ params,
                                               const _Float16* __restrict__ ampsT,
                                               float* __restrict__ out) {
    __shared__ _Float16 A_lds[2][BLK_P * A_STRIDE];     // double-buffered basis tile (2x5KB)
    __shared__ float pos_sh[BLK_P];

    const int p0   = blockIdx.x * BLK_P;
    const int b    = blockIdx.y;
    const int tid  = threadIdx.x;
    const int lane = tid & 31;
    const int wave = tid >> 5;
    const int p_sub = wave & 1;                         // 2 wave-rows of 32 P
    const int d_sub = wave >> 1;                        // 4 wave-cols of 64 D

    if (tid < BLK_P) pos_sh[tid] = pos[b * PP + p0 + tid];

    v8f acc[2][4];
    const v8f vzero = {0.f, 0.f, 0.f, 0.f, 0.f, 0.f, 0.f, 0.f};
#pragma unroll
    for (int i = 0; i < 2; ++i)
#pragma unroll
        for (int j = 0; j < 4; ++j) acc[i][j] = vzero;

    const int m   = lane & 15;
    const int kbA = (lane >> 4) * 8;                    // A frag K base (ISA 16-bit A layout)
    const int kbB = (lane >> 4) * 8;                    // B frag K base (same K order as A)

    const _Float16* Bbase[4];
#pragma unroll
    for (int j = 0; j < 4; ++j) {
        int d = d_sub * 64 + j * 16 + m;
        Bbase[j] = ampsT + ((size_t)(b * DD + d)) * SS;
    }

    // A-tile generation mapping: thread t handles s = k0 + (t&31) for 8 p values
    const int kk = tid & 31;
    const int pr = tid >> 5;
    const float* prm_base = params + ((size_t)b * SS) * PSTRIDE;

    __syncthreads();                                    // pos_sh visible
    gen_A(A_lds[0], prm_base, pos_sh, 0, kk, pr);       // prologue tile
    __syncthreads();

    for (int k0 = 0; k0 < SS; k0 += KSTEP) {
        const int cur = (k0 >> 5) & 1;
        _Float16* curbuf = A_lds[cur];
        _Float16* nxtbuf = A_lds[cur ^ 1];

        v16h afrag[2], bfrag[4];
#pragma unroll
        for (int i = 0; i < 2; ++i) {                   // A: lanes 0-15 K 0..7|16..23, lanes 16-31 K 8..15|24..31
            int prow = p_sub * 32 + i * 16 + m;
            const _Float16* ap = &curbuf[prow * A_STRIDE + kbA];
            v8h lo = *(const v8h*)(ap);
            v8h hi = *(const v8h*)(ap + 16);
#pragma unroll
            for (int e = 0; e < 8; ++e) { afrag[i][e] = lo[e]; afrag[i][e + 8] = hi[e]; }
        }
#pragma unroll
        for (int j = 0; j < 4; ++j) {                   // B: lanes 0-15 K 0..7|16..23, lanes 16-31 K 8..15|24..31
            const _Float16* bp = Bbase[j] + k0 + kbB;
            v8h lo = *(const v8h*)(bp);
            v8h hi = *(const v8h*)(bp + 16);
#pragma unroll
            for (int e = 0; e < 8; ++e) { bfrag[j][e] = lo[e]; bfrag[j][e + 8] = hi[e]; }
            __builtin_prefetch((const void*)(bp + KSTEP), 0, 3);  // global_prefetch_b8, next K-step
        }

        if (k0 + KSTEP < SS)                            // generate next tile; VALU/trans co-executes
            gen_A(nxtbuf, prm_base, pos_sh, k0 + KSTEP, kk, pr);  // with the XDL WMMA pipe

#pragma unroll
        for (int i = 0; i < 2; ++i)
#pragma unroll
            for (int j = 0; j < 4; ++j)
                acc[i][j] = __builtin_amdgcn_wmma_f32_16x16x32_f16(
                    false, afrag[i], false, bfrag[j], (short)0, acc[i][j], false, false);
        __syncthreads();                                // nxt tile written; cur tile free
    }

    // C/D layout: reg r -> (M = r + 8*(lane>>4), N = lane&15)
    const int mc = (lane >> 4) * 8;
    const int n  = lane & 15;
#pragma unroll
    for (int i = 0; i < 2; ++i)
#pragma unroll
        for (int j = 0; j < 4; ++j) {
            int dg = d_sub * 64 + j * 16 + n;
#pragma unroll
            for (int r = 0; r < 8; ++r) {
                int pg = p0 + p_sub * 32 + i * 16 + mc + r;
                out[((size_t)(b * PP + pg)) * DD + dg] = acc[i][j][r];
            }
        }
}

// ---------------------------------------------------------------------------

extern "C" void kernel_launch(void* const* d_in, const int* in_sizes, int n_in,
                              void* d_out, int out_size, void* d_ws, size_t ws_size,
                              hipStream_t stream) {
    const int*   bytes = (const int*)d_in[0];
    const float* pos   = (const float*)d_in[1];
    const float* emb   = (const float*)d_in[2];
    const float* lsh   = (const float*)d_in[3];
    const float* w1 = (const float*)d_in[4];
    const float* b1 = (const float*)d_in[5];
    const float* w2 = (const float*)d_in[6];
    const float* b2 = (const float*)d_in[7];
    const float* w3 = (const float*)d_in[8];
    const float* b3 = (const float*)d_in[9];
    const float* msw = (const float*)d_in[10];
    const float* sw  = (const float*)d_in[11];

    char* ws = (char*)d_ws;
    float*    h1    = (float*)(ws);                      //  8 MB: [B][128][S]
    float*    h2    = (float*)(ws + (8u  << 20));        //  4 MB: [B][64][S]
    float*    ctx   = (float*)(ws + (12u << 20));        // 64 KB: [B][S]
    float*    prm   = (float*)(ws + (13u << 20));        //  1 MB: [B][S][16]
    _Float16* ampsT = (_Float16*)(ws + (14u << 20));     //  8 MB: [B][D][S] fp16

    k_conv1 <<<(BB * 128 * SS) / 256, 256, 0, stream>>>(bytes, w1, b1, h1);
    k_conv2 <<<(BB * 64  * SS) / 256, 256, 0, stream>>>(h1, w2, b2, h2);
    k_conv3 <<<(BB * SS) / 256,        256, 0, stream>>>(h2, w3, b3, ctx);
    k_params<<<(BB * SS) / 256,        256, 0, stream>>>(bytes, lsh, msw, sw, prm);
    k_amps  <<<(BB * DD * SS) / 256,   256, 0, stream>>>(bytes, emb, ctx, ampsT);

    dim3 grid(PP / BLK_P, BB);                           // (32, 8) blocks, full D per block
    k_field<<<grid, 256, 0, stream>>>(pos, prm, ampsT, (float*)d_out);
}
